// NTT_5162550690116
// MI455X (gfx1250) — matrix operand order repaired and checked by reading, not compiled, not run on hardware
//
#include <hip/hip_runtime.h>
#include <math.h>

// ---------------------------------------------------------------------------
// CDNA5 (gfx1250) forward pass for the 3D-CNN + ViT encoder + transformer
// decoder reference. All GEMM-shaped work runs through
// v_wmma_f32_16x16x32_bf16 (bf16 in, f32 accumulate), 16x64 tile per wave.
// All weights are pre-transposed on device to [N,K] so every WMMA fragment
// (A and B) is loaded with contiguous global_load_b128 - no strided gathers
// anywhere except the inherently scattered conv input (clause-batched).
// ---------------------------------------------------------------------------

typedef __attribute__((ext_vector_type(16))) __bf16 v16bf;
typedef __attribute__((ext_vector_type(8)))  float  v8f;

#define HEADS 8
#define DH    64
#define DIMM  512
#define NT    4      // 16x16 sub-tiles per wave along N

// WMMA fragment index maps (CDNA5 ISA 7.12.2, wave32):
//  A (16x32 bf16): lane<16 -> row M=lane, halfs cover K {0..7,16..23};
//                  lane>=16 -> same rows, K {8..15,24..31}.
//  B (32x16 bf16): lane<16 -> col N=lane, K {0..7,16..23}; lane>=16 -> K {8..15,24..31}.
//  C/D (16x16 f32): vgpr r -> M = r + 8*(lane>=16), N = lane&15.
__device__ __forceinline__ int a_kmap(int e, int lane) {
  return (e & 7) + ((e >> 3) << 4) + ((lane >> 4) << 3);
}

__device__ __forceinline__ v16bf pack16(float4 u0, float4 u1, float4 u2, float4 u3) {
  v16bf r;
  r[0]  = (__bf16)u0.x; r[1]  = (__bf16)u0.y; r[2]  = (__bf16)u0.z; r[3]  = (__bf16)u0.w;
  r[4]  = (__bf16)u1.x; r[5]  = (__bf16)u1.y; r[6]  = (__bf16)u1.z; r[7]  = (__bf16)u1.w;
  r[8]  = (__bf16)u2.x; r[9]  = (__bf16)u2.y; r[10] = (__bf16)u2.z; r[11] = (__bf16)u2.w;
  r[12] = (__bf16)u3.x; r[13] = (__bf16)u3.y; r[14] = (__bf16)u3.z; r[15] = (__bf16)u3.w;
  return r;
}

// A fragment from a row contiguous in K: segments [off+8h, +8) and [off+16+8h, +8).
__device__ __forceinline__ v16bf load_a_idx(const float* __restrict__ base, int off, int half) {
  int o0 = off + half * 8;
  int o1 = off + 16 + half * 8;
  return pack16(*(const float4*)(base + o0), *(const float4*)(base + o0 + 4),
                *(const float4*)(base + o1), *(const float4*)(base + o1 + 4));
}

// B fragment contiguous in K per row: segments [off+8h, +8) and [off+16+8h, +8).
__device__ __forceinline__ v16bf load_k16_idx(const float* __restrict__ base, int off, int half) {
  int o0 = off + half * 8;
  int o1 = off + 16 + half * 8;
  return pack16(*(const float4*)(base + o0), *(const float4*)(base + o0 + 4),
                *(const float4*)(base + o1), *(const float4*)(base + o1 + 4));
}

// ---------------------------------------------------------------------------
// GEMM with pre-transposed B: C[M,N] = A[M,K] @ BT[N,K]^T (+bias) (+resid)
// One wave computes a 16x64 tile; every fragment load is b128.
// ---------------------------------------------------------------------------
__global__ void k_gemm_bt(const float* __restrict__ A, int lda,
                          const float* __restrict__ BT,
                          const float* __restrict__ bias,
                          const float* __restrict__ resid, int ldr,
                          float* __restrict__ C, int ldc,
                          int M, int N, int K)
{
  int lane = threadIdx.x;
  int l15 = lane & 15, half = lane >> 4;
  int mbase = blockIdx.x * 16;
  int nbase = blockIdx.y * (16 * NT);
  int rowc = min(mbase + l15, M - 1);
  int aOff = rowc * lda;
  int colc[NT], bOff[NT];
#pragma unroll
  for (int t = 0; t < NT; ++t) {
    colc[t] = min(nbase + t * 16 + l15, N - 1);
    bOff[t] = colc[t] * K;
  }

  v8f acc[NT] = {};
  int Kmain = K & ~31;
  for (int k0 = 0; k0 < Kmain; k0 += 32) {
    v16bf a = load_a_idx(A, aOff + k0, half);
#pragma unroll
    for (int t = 0; t < NT; ++t) {
      v16bf b = load_k16_idx(BT, bOff[t] + k0, half);
      acc[t] = __builtin_amdgcn_wmma_f32_16x16x32_bf16(false, a, false, b,
                                                       (short)0, acc[t], false, false);
    }
  }
  if (Kmain < K) {            // K tail (unused by current shapes; kept for safety)
    v16bf a;
#pragma unroll
    for (int e = 0; e < 16; ++e) {
      int ka = Kmain + a_kmap(e, lane);
      float av = A[aOff + min(ka, K - 1)];
      a[e] = (__bf16)(ka < K ? av : 0.f);
    }
#pragma unroll
    for (int t = 0; t < NT; ++t) {
      v16bf b;
#pragma unroll
      for (int e = 0; e < 16; ++e) {
        int kb = Kmain + a_kmap(e, lane);
        float bv = BT[bOff[t] + min(kb, K - 1)];
        b[e] = (__bf16)(kb < K ? bv : 0.f);
      }
      acc[t] = __builtin_amdgcn_wmma_f32_16x16x32_bf16(false, a, false, b,
                                                       (short)0, acc[t], false, false);
    }
  }
#pragma unroll
  for (int t = 0; t < NT; ++t) {
#pragma unroll
    for (int r = 0; r < 8; ++r) {
      int mo = mbase + r + (half << 3);
      int no = nbase + t * 16 + l15;
      if (mo < M && no < N) {
        float v = acc[t][r];
        if (bias)  v += bias[no];
        if (resid) v += resid[mo * ldr + no];
        C[mo * ldc + no] = v;
      }
    }
  }
}

// dst[n*K + k] = src[k*N + n]   (weight [K,N] -> [N,K])
__global__ void k_transpose(const float* __restrict__ src, float* __restrict__ dst,
                            int K, int N)
{
  int idx = blockIdx.x * blockDim.x + threadIdx.x;
  if (idx < K * N) {
    int k = idx / N, n = idx - k * N;
    dst[n * K + k] = src[idx];
  }
}

// vT[bh*64 + d][j] = qkv[(b*n + j)*1536 + 1024 + h*64 + d]
__global__ void k_transpose_v(const float* __restrict__ qkv, float* __restrict__ vT, int n)
{
  int idx = blockIdx.x * blockDim.x + threadIdx.x;
  if (idx < 16 * DH * n) {
    int bh = idx / (DH * n);
    int r  = idx - bh * (DH * n);
    int d  = r / n, j = r - d * n;
    int b = bh / HEADS, h = bh - b * HEADS;
    vT[idx] = qkv[(b * n + j) * (3 * DIMM) + 2 * DIMM + h * DH + d];
  }
}

// ---------------------------------------------------------------------------
// Implicit-GEMM conv3d (NCDHW, OIDHW weights, pad=(KS-1)/2).
// M = Do*Ho*Wo voxels (per batch), N = Co, K = Ci*KS^3.
// Weights [Co, K] contiguous in K per output channel -> b128 loads.
// Input gather: offsets+masks computed, then 16 loads (clause), then pack.
// Conv bias omitted: every conv feeds InstanceNorm which cancels it exactly.
// ---------------------------------------------------------------------------
template<int KS>
__global__ void k_conv3d(const float* __restrict__ in, const float* __restrict__ w,
                         float* __restrict__ out,
                         int Ci, int Co, int Di, int Hi, int Wi,
                         int Do, int Ho, int Wo, int stride)
{
  const int kvol = KS * KS * KS;
  const int pad  = (KS - 1) / 2;
  int lane = threadIdx.x;
  int l15 = lane & 15, half = lane >> 4;
  int n = blockIdx.z;
  int Msp = Do * Ho * Wo;
  int mbase = blockIdx.x * 16;
  int cobase = blockIdx.y * (16 * NT);
  int mm = min(mbase + l15, Msp - 1);
  int oz = mm / (Ho * Wo);
  int rm = mm - oz * (Ho * Wo);
  int oy = rm / Wo;
  int ox = rm - oy * Wo;
  int iz0 = oz * stride - pad, iy0 = oy * stride - pad, ix0 = ox * stride - pad;
  int K = Ci * kvol;
  int inOff = n * Ci * Di * Hi * Wi;       // < 2^31 for all stages
  int coc[NT], wOff[NT];
#pragma unroll
  for (int t = 0; t < NT; ++t) {
    coc[t] = min(cobase + t * 16 + l15, Co - 1);
    wOff[t] = coc[t] * K;
  }

  v8f acc[NT] = {};
  int Kmain = K & ~31;
  for (int k0 = 0; k0 < Kmain; k0 += 32) {
    int offs[16];
    unsigned msk = 0;
#pragma unroll
    for (int e = 0; e < 16; ++e) {
      int ka = k0 + a_kmap(e, lane);
      int ci = ka / kvol;
      int r  = ka - ci * kvol;
      int kz = r / (KS * KS);
      int r2 = r - kz * KS * KS;
      int ky = r2 / KS;
      int kx = r2 - ky * KS;
      int iz = iz0 + kz, iy = iy0 + ky, ix = ix0 + kx;
      bool inb = (unsigned)iz < (unsigned)Di && (unsigned)iy < (unsigned)Hi &&
                 (unsigned)ix < (unsigned)Wi;
      msk |= (inb ? 1u : 0u) << e;
      int izc = min(max(iz, 0), Di - 1);
      int iyc = min(max(iy, 0), Hi - 1);
      int ixc = min(max(ix, 0), Wi - 1);
      offs[e] = inOff + ((ci * Di + izc) * Hi + iyc) * Wi + ixc;
    }
    float av[16];
#pragma unroll
    for (int e = 0; e < 16; ++e) av[e] = in[offs[e]];
    v16bf a;
#pragma unroll
    for (int e = 0; e < 16; ++e)
      a[e] = (__bf16)((msk >> e) & 1u ? av[e] : 0.f);
#pragma unroll
    for (int t = 0; t < NT; ++t) {
      v16bf b = load_k16_idx(w, wOff[t] + k0, half);
      acc[t] = __builtin_amdgcn_wmma_f32_16x16x32_bf16(false, a, false, b,
                                                       (short)0, acc[t], false, false);
    }
  }
  if (Kmain < K) {            // only hit by the first conv (K = 27)
    int offs[16];
    unsigned msk = 0;
#pragma unroll
    for (int e = 0; e < 16; ++e) {
      int ka = Kmain + a_kmap(e, lane);
      int kac = min(ka, K - 1);
      int ci = kac / kvol;
      int r  = kac - ci * kvol;
      int kz = r / (KS * KS);
      int r2 = r - kz * KS * KS;
      int ky = r2 / KS;
      int kx = r2 - ky * KS;
      int iz = iz0 + kz, iy = iy0 + ky, ix = ix0 + kx;
      bool inb = (ka < K) && (unsigned)iz < (unsigned)Di &&
                 (unsigned)iy < (unsigned)Hi && (unsigned)ix < (unsigned)Wi;
      msk |= (inb ? 1u : 0u) << e;
      int izc = min(max(iz, 0), Di - 1);
      int iyc = min(max(iy, 0), Hi - 1);
      int ixc = min(max(ix, 0), Wi - 1);
      offs[e] = inOff + ((ci * Di + izc) * Hi + iyc) * Wi + ixc;
    }
    float av[16];
#pragma unroll
    for (int e = 0; e < 16; ++e) av[e] = in[offs[e]];
    v16bf a;
#pragma unroll
    for (int e = 0; e < 16; ++e)
      a[e] = (__bf16)((msk >> e) & 1u ? av[e] : 0.f);
#pragma unroll
    for (int t = 0; t < NT; ++t) {
      float bv[16];
#pragma unroll
      for (int e = 0; e < 16; ++e)
        bv[e] = w[wOff[t] + min(Kmain + a_kmap(e, lane), K - 1)];
      v16bf b;
#pragma unroll
      for (int e = 0; e < 16; ++e)
        b[e] = (__bf16)((Kmain + a_kmap(e, lane)) < K ? bv[e] : 0.f);
      acc[t] = __builtin_amdgcn_wmma_f32_16x16x32_bf16(false, a, false, b,
                                                       (short)0, acc[t], false, false);
    }
  }
#pragma unroll
  for (int t = 0; t < NT; ++t) {
#pragma unroll
    for (int r = 0; r < 8; ++r) {
      int mo = mbase + r + (half << 3);
      int no = cobase + t * 16 + l15;
      if (mo < Msp && no < Co)
        out[(n * Co + no) * Msp + mo] = acc[t][r];
    }
  }
}

// ---------------------------------------------------------------------------
// InstanceNorm3d(affine) + optional residual add + activation.
// act: 0=none, 1=LeakyReLU(0.01), 2=ELU(alpha=0.2). One block per (n,c).
// ---------------------------------------------------------------------------
__global__ void k_instnorm_act(const float* __restrict__ x, const float* __restrict__ g,
                               const float* __restrict__ be, const float* __restrict__ resid,
                               float* __restrict__ out, int C, int S, int act)
{
  __shared__ float sh[256];
  int nc = blockIdx.x;
  int c  = nc % C;
  int base = nc * S;                       // < 2^31
  int t = threadIdx.x;
  float s = 0.f, s2 = 0.f;
  for (int i = t; i < S; i += 256) { float v = x[base + i]; s += v; s2 += v * v; }
  sh[t] = s; __syncthreads();
  for (int st = 128; st > 0; st >>= 1) { if (t < st) sh[t] += sh[t + st]; __syncthreads(); }
  float sum = sh[0]; __syncthreads();
  sh[t] = s2; __syncthreads();
  for (int st = 128; st > 0; st >>= 1) { if (t < st) sh[t] += sh[t + st]; __syncthreads(); }
  float sum2 = sh[0]; __syncthreads();
  float mean = sum / S;
  float rstd = rsqrtf(sum2 / S - mean * mean + 1e-5f);
  float gg = g[c], bb = be[c];
  for (int i = t; i < S; i += 256) {
    float v = (x[base + i] - mean) * rstd * gg + bb;
    if (resid) v += resid[base + i];
    if (act == 1)      v = v > 0.f ? v : 0.01f * v;
    else if (act == 2) v = v > 0.f ? v : 0.2f * expm1f(v);
    out[base + i] = v;
  }
}

// LayerNorm over last dim D; one block per row.
__global__ void k_layernorm(const float* __restrict__ x, const float* __restrict__ g,
                            const float* __restrict__ b, float* __restrict__ out, int D)
{
  __shared__ float sh[256];
  int base = blockIdx.x * D;
  int t = threadIdx.x;
  float s = 0.f, s2 = 0.f;
  for (int i = t; i < D; i += 256) { float v = x[base + i]; s += v; s2 += v * v; }
  sh[t] = s; __syncthreads();
  for (int st = 128; st > 0; st >>= 1) { if (t < st) sh[t] += sh[t + st]; __syncthreads(); }
  float sum = sh[0]; __syncthreads();
  sh[t] = s2; __syncthreads();
  for (int st = 128; st > 0; st >>= 1) { if (t < st) sh[t] += sh[t + st]; __syncthreads(); }
  float sum2 = sh[0]; __syncthreads();
  float mean = sum / D;
  float rstd = rsqrtf(sum2 / D - mean * mean + 1e-5f);
  for (int i = t; i < D; i += 256)
    out[base + i] = (x[base + i] - mean) * rstd * g[i] + b[i];
}

// scores[bh,i,j] = scale * sum_d q(i,d)*k(j,d); grid=(n/16, n/64, B*HEADS)
// Q rows and K rows are contiguous (DH floats) -> all b128 loads; K dim = 64.
__global__ void k_attn_scores(const float* __restrict__ qkv, int ld, int qoff, int koff,
                              int n, float scale, float* __restrict__ sc)
{
  int lane = threadIdx.x;
  int l15 = lane & 15, half = lane >> 4;
  int bh = blockIdx.z;
  int b = bh / HEADS, h = bh % HEADS;
  int ibase = blockIdx.x * 16, jbase = blockIdx.y * (16 * NT);
  int ic = min(ibase + l15, n - 1);
  int qOff = b * n * ld + qoff + h * DH + ic * ld;
  int kBase = b * n * ld + koff + h * DH;
  int kOff[NT];
#pragma unroll
  for (int t = 0; t < NT; ++t)
    kOff[t] = kBase + min(jbase + t * 16 + l15, n - 1) * ld;
  v8f acc[NT] = {};
#pragma unroll
  for (int k0 = 0; k0 < DH; k0 += 32) {
    v16bf a = load_a_idx(qkv, qOff + k0, half);
#pragma unroll
    for (int t = 0; t < NT; ++t) {
      v16bf b = load_k16_idx(qkv, kOff[t] + k0, half);
      acc[t] = __builtin_amdgcn_wmma_f32_16x16x32_bf16(false, a, false, b,
                                                       (short)0, acc[t], false, false);
    }
  }
#pragma unroll
  for (int t = 0; t < NT; ++t) {
#pragma unroll
    for (int r = 0; r < 8; ++r) {
      int mo = ibase + r + (half << 3);
      int no = jbase + t * 16 + l15;
      if (mo < n && no < n)
        sc[(bh * n + mo) * n + no] = acc[t][r] * scale;
    }
  }
}

// o[b,i,h*DH+d] = sum_j probs(i,j) * vT[bh,d,j]; grid=(n/16, 1, B*HEADS)
// Both operands contiguous in K (= j) -> all b128 loads.
__global__ void k_attn_out(const float* __restrict__ pr, const float* __restrict__ vT,
                           int n, float* __restrict__ o)
{
  int lane = threadIdx.x;
  int l15 = lane & 15, half = lane >> 4;
  int bh = blockIdx.z;
  int b = bh / HEADS, h = bh % HEADS;
  int ibase = blockIdx.x * 16;
  int ic = min(ibase + l15, n - 1);
  int pOff = bh * n * n + ic * n;
  int dOff[NT];
#pragma unroll
  for (int t = 0; t < NT; ++t)
    dOff[t] = (bh * DH + t * 16 + l15) * n;
  v8f acc[NT] = {};
  int Kmain = n & ~31;
  for (int k0 = 0; k0 < Kmain; k0 += 32) {
    v16bf a = load_a_idx(pr, pOff + k0, half);
#pragma unroll
    for (int t = 0; t < NT; ++t) {
      v16bf bfr = load_k16_idx(vT, dOff[t] + k0, half);
      acc[t] = __builtin_amdgcn_wmma_f32_16x16x32_bf16(false, a, false, bfr,
                                                       (short)0, acc[t], false, false);
    }
  }
  if (Kmain < n) {            // n tail (encoder n = 36)
    v16bf a;
#pragma unroll
    for (int e = 0; e < 16; ++e) {
      int ka = Kmain + a_kmap(e, lane);
      float av = pr[pOff + min(ka, n - 1)];
      a[e] = (__bf16)(ka < n ? av : 0.f);
    }
#pragma unroll
    for (int t = 0; t < NT; ++t) {
      v16bf bfr;
#pragma unroll
      for (int e = 0; e < 16; ++e) {
        int kb = Kmain + a_kmap(e, lane);
        float bv = vT[dOff[t] + min(kb, n - 1)];
        bfr[e] = (__bf16)(kb < n ? bv : 0.f);
      }
      acc[t] = __builtin_amdgcn_wmma_f32_16x16x32_bf16(false, a, false, bfr,
                                                       (short)0, acc[t], false, false);
    }
  }
#pragma unroll
  for (int t = 0; t < NT; ++t) {
#pragma unroll
    for (int r = 0; r < 8; ++r) {
      int mo = ibase + r + (half << 3);
      if (mo < n)
        o[(b * n + mo) * DIMM + h * DH + t * 16 + l15] = acc[t][r];
    }
  }
}

// softmax per row; causal=1 adds causal + pad-key masking (padf[b*n+j]==0 -> masked)
__global__ void k_softmax(float* __restrict__ sc, int n, int causal,
                          const float* __restrict__ padf)
{
  __shared__ float sh[128];
  int row = blockIdx.x;            // bh*n + i
  int i = row % n;
  int b = row / (n * HEADS);
  int base = row * n;
  int t = threadIdx.x;
  float mx = -1e30f;
  for (int j = t; j < n; j += 128) {
    bool msk = causal && (j > i || (padf && padf[b * n + j] == 0.f));
    if (!msk) mx = fmaxf(mx, sc[base + j]);
  }
  sh[t] = mx; __syncthreads();
  for (int st = 64; st > 0; st >>= 1) { if (t < st) sh[t] = fmaxf(sh[t], sh[t + st]); __syncthreads(); }
  mx = sh[0]; __syncthreads();
  float s = 0.f;
  for (int j = t; j < n; j += 128) {
    bool msk = causal && (j > i || (padf && padf[b * n + j] == 0.f));
    float e = msk ? 0.f : expf(sc[base + j] - mx);
    sc[base + j] = e; s += e;
  }
  sh[t] = s; __syncthreads();
  for (int st = 64; st > 0; st >>= 1) { if (t < st) sh[t] += sh[t + st]; __syncthreads(); }
  s = sh[0];
  float inv = s > 0.f ? 1.f / s : 0.f;
  for (int j = t; j < n; j += 128) sc[base + j] *= inv;
}

__global__ void k_gelu(float* __restrict__ x, int nelem)
{
  int i = blockIdx.x * blockDim.x + threadIdx.x;
  if (i < nelem) { float v = x[i]; x[i] = 0.5f * v * (1.f + erff(v * 0.70710678118654752f)); }
}

__global__ void k_padflag(const float* __restrict__ x, float* __restrict__ padf)
{
  int i = threadIdx.x;           // 256 rows
  if (i < 256) padf[i] = (x[i * 32 + 3] > 0.f) ? 1.f : 0.f;
}

__global__ void k_bcast_add(float* __restrict__ xd, const float* __restrict__ ca)
{
  int idx = blockIdx.x * blockDim.x + threadIdx.x;   // 256*512
  if (idx < 256 * DIMM) {
    int row = idx / DIMM, c = idx % DIMM;
    xd[idx] += ca[(row / 128) * DIMM + c];
  }
}

__global__ void k_mean_tokens(const float* __restrict__ xe, float* __restrict__ mem, int ntok)
{
  int i = blockIdx.x * blockDim.x + threadIdx.x;     // 2*512
  if (i < 2 * DIMM) {
    int b = i / DIMM, c = i % DIMM;
    float s = 0.f;
    for (int t = 0; t < ntok; ++t) s += xe[(b * ntok + t) * DIMM + c];
    mem[i] = s / (float)ntok;
  }
}

__global__ void k_add_pe1d(float* __restrict__ xd, int n)
{
  int idx = blockIdx.x * blockDim.x + threadIdx.x;   // rows*512
  if (idx < 256 * DIMM) {
    int row = idx / DIMM, c = idx % DIMM;
    int pos = row % n;
    int i = (c < 256) ? c : c - 256;
    float om = powf(10000.f, -(float)i / 255.f);
    float t = (float)pos * om;
    xd[idx] += (c < 256) ? sinf(t) : cosf(t);
  }
}

__global__ void k_add_pe3d(float* __restrict__ xe)
{
  int idx = blockIdx.x * blockDim.x + threadIdx.x;   // 72*512
  if (idx < 72 * DIMM) {
    int row = idx / DIMM, c = idx % DIMM;
    int t = row % 36;
    int z = t / 9, y = (t / 3) % 3, xx = t % 3;
    float v = 0.f;
    if (c < 510) {
      int axis = c / 170, within = c % 170;
      int coord = (axis == 0) ? xx : (axis == 1 ? y : z);
      int i = (within < 85) ? within : within - 85;
      float om = powf(10000.f, -(float)i / 84.f);
      float arg = (float)coord * om;
      v = (within < 85) ? sinf(arg) : cosf(arg);
    }
    xe[idx] += v;
  }
}

// img3 [2,256,8,12,12] -> xp [2,36,8192] with (p1 p2 p3 c) inner ordering
__global__ void k_patchify(const float* __restrict__ img3, float* __restrict__ xp)
{
  int idx = blockIdx.x * blockDim.x + threadIdx.x;   // 2*36*8192
  if (idx < 2 * 36 * 8192) {
    int b = idx / (36 * 8192);
    int r = idx % (36 * 8192);
    int t = r / 8192, f = r % 8192;
    int z = t / 9, y = (t / 3) % 3, xx = t % 3;
    int p1 = f / 4096, p2 = (f / 1024) % 4, p3 = (f / 256) % 4, c = f % 256;
    int dz = z * 2 + p1, wy = y * 4 + p2, hx = xx * 4 + p3;
    xp[idx] = img3[((b * 256 + c) * 8 + dz) * 144 + wy * 12 + hx];
  }
}

// ---------------------------------------------------------------------------
// Host orchestration
// ---------------------------------------------------------------------------
extern "C" void kernel_launch(void* const* d_in, const int* in_sizes, int n_in,
                              void* d_out, int out_size, void* d_ws, size_t ws_size,
                              hipStream_t stream)
{
  (void)in_sizes; (void)n_in; (void)out_size;
  int ii = 0;
  auto nxt = [&]() { return (const float*)d_in[ii++]; };

  // jax pytree leaf order: dicts sorted by key, lists in order.
  // top-level sorted: img < params < x; params sorted: dec<downs<enc<pre<proj.
  const float* img = nxt();

  struct LN  { const float* b; const float* g; };
  struct CAp { const float* kv; LN ln; const float* out; const float* q; };
  struct FFp { const float* b1; const float* b2; LN ln; const float* w1; const float* w2; };
  struct SAp { LN ln; const float* out; const float* qkv; };
  struct CB  { const float* b; const float* be; const float* g; const float* w; };

  // dec: be2, layers(ca,ff,sa), ln1, ln2, we
  const float* dec_be2 = nxt();
  CAp dca[6]; FFp dff[6]; SAp dsa[6];
  for (int l = 0; l < 6; ++l) {
    dca[l].kv = nxt(); dca[l].ln.b = nxt(); dca[l].ln.g = nxt(); dca[l].out = nxt(); dca[l].q = nxt();
    dff[l].b1 = nxt(); dff[l].b2 = nxt(); dff[l].ln.b = nxt(); dff[l].ln.g = nxt(); dff[l].w1 = nxt(); dff[l].w2 = nxt();
    dsa[l].ln.b = nxt(); dsa[l].ln.g = nxt(); dsa[l].out = nxt(); dsa[l].qkv = nxt();
  }
  LN dec_ln1; dec_ln1.b = nxt(); dec_ln1.g = nxt();
  LN dec_ln2; dec_ln2.b = nxt(); dec_ln2.g = nxt();
  const float* dec_we = nxt();

  // downs: 3 x {l1,l2,sc}, each cb sorted b,be,g,w
  CB dn[3][3];
  for (int d = 0; d < 3; ++d)
    for (int j = 0; j < 3; ++j) { dn[d][j].b = nxt(); dn[d][j].be = nxt(); dn[d][j].g = nxt(); dn[d][j].w = nxt(); }

  // enc: bp, layers(attn,ff), ln1, ln2, wp
  const float* enc_bp = nxt();
  SAp esa[6]; FFp eff[6];
  for (int l = 0; l < 6; ++l) {
    esa[l].ln.b = nxt(); esa[l].ln.g = nxt(); esa[l].out = nxt(); esa[l].qkv = nxt();
    eff[l].b1 = nxt(); eff[l].b2 = nxt(); eff[l].ln.b = nxt(); eff[l].ln.g = nxt(); eff[l].w1 = nxt(); eff[l].w2 = nxt();
  }
  LN enc_ln1; enc_ln1.b = nxt(); enc_ln1.g = nxt();
  LN enc_ln2; enc_ln2.b = nxt(); enc_ln2.g = nxt();
  const float* enc_wp = nxt();

  // pre: 2 conv blocks; proj: b, w; then x
  CB pre[2];
  for (int j = 0; j < 2; ++j) { pre[j].b = nxt(); pre[j].be = nxt(); pre[j].g = nxt(); pre[j].w = nxt(); }
  const float* proj_b = nxt();
  const float* proj_w = nxt();
  const float* x = nxt();

  // ---- workspace: two full-res ping-pong buffers ----
  const size_t F0 = (size_t)2 * 32 * 64 * 96 * 96;   // 37,748,736 floats
  if (ws_size < 2 * F0 * sizeof(float)) return;
  float* BUF0 = (float*)d_ws;
  float* BUF1 = BUF0 + F0;

  dim3 wave(32);
  auto gemmt = [&](const float* A, int lda, const float* BT, int N_,
                   const float* bias, const float* resid, float* C, int M_, int K_) {
    dim3 g((M_ + 15) / 16, (N_ + 63) / 64);
    k_gemm_bt<<<g, wave, 0, stream>>>(A, lda, BT, bias, resid, N_, C, N_, M_, N_, K_);
  };
  auto transp = [&](const float* src, float* dst, int K_, int N_) {
    int tot = K_ * N_;
    k_transpose<<<dim3((tot + 255) / 256), dim3(256), 0, stream>>>(src, dst, K_, N_);
  };
  auto lnorm = [&](const float* xin, LN p, float* o, int rows, int D) {
    k_layernorm<<<dim3(rows), dim3(256), 0, stream>>>(xin, p.g, p.b, o, D);
  };
  auto inorm = [&](const float* xin, const CB& cb, const float* resid, float* o,
                   int C, int S, int act) {
    k_instnorm_act<<<dim3(2 * C), dim3(256), 0, stream>>>(xin, cb.g, cb.be, resid, o, C, S, act);
  };
  auto conv3 = [&](const float* in_, const float* w_, float* out_, int Ci, int Co,
                   int Di, int Hi, int Wi, int s) {
    int Do = (Di - 1) / s + 1, Ho = (Hi - 1) / s + 1, Wo = (Wi - 1) / s + 1;
    int M = Do * Ho * Wo;
    dim3 g((M + 15) / 16, (Co + 63) / 64, 2);
    k_conv3d<3><<<g, wave, 0, stream>>>(in_, w_, out_, Ci, Co, Di, Hi, Wi, Do, Ho, Wo, s);
  };

  // ============================ pre layers ============================
  conv3(img, pre[0].w, BUF0, 1, 32, 64, 96, 96, 1);
  inorm(BUF0, pre[0], nullptr, BUF0, 32, 64 * 96 * 96, 1);      // LeakyReLU(0.01)
  conv3(BUF0, pre[1].w, BUF1, 32, 32, 64, 96, 96, 1);
  inorm(BUF1, pre[1], nullptr, BUF1, 32, 64 * 96 * 96, 1);

  // ========================= down blocks =========================
  const float* din = BUF1;
  float* pools[3] = { BUF0, BUF1, BUF0 };
  int Ci = 32, Di = 64, Hi = 96, Wi = 96;
  int Cos[3] = { 64, 128, 256 };
  float* cur = nullptr;
  for (int d = 0; d < 3; ++d) {
    int Co = Cos[d];
    int Do = Di / 2, Ho = Hi / 2, Wo = Wi / 2;
    int S = Do * Ho * Wo;
    size_t bs = (size_t)2 * Co * S;
    float* t1 = pools[d];
    float* t2 = pools[d] + bs;
    float* t3 = pools[d] + 2 * bs;
    conv3(din, dn[d][0].w, t1, Ci, Co, Di, Hi, Wi, 2);
    inorm(t1, dn[d][0], nullptr, t1, Co, S, 2);                 // ELU(0.2)
    conv3(t1, dn[d][1].w, t2, Co, Co, Do, Ho, Wo, 1);
    inorm(t2, dn[d][1], nullptr, t2, Co, S, 0);                 // no act
    {   // 1x1x1 stride-2 shortcut
      dim3 g((S + 15) / 16, (Co + 63) / 64, 2);
      k_conv3d<1><<<g, wave, 0, stream>>>(din, dn[d][2].w, t3, Ci, Co,
                                          Di, Hi, Wi, Do, Ho, Wo, 2);
    }
    inorm(t3, dn[d][2], t2, t3, Co, S, 2);                      // elu(norm(sc)+norm(l2))
    din = t3; cur = t3;
    Ci = Co; Di = Do; Hi = Ho; Wi = Wo;
  }
  // cur = img3 [2,256,8,12,12] in BUF0

  // ========================= ViT encoder =========================
  float* xp  = BUF1;                    // [72, 8192]
  float* xn8 = BUF1 + 589824;           // [72, 8192]
  k_patchify<<<dim3((2 * 36 * 8192 + 255) / 256), dim3(256), 0, stream>>>(cur, xp);
  lnorm(xp, enc_ln1, xn8, 72, 8192);

  // ---- transposed weights: enc side in BUF1, dec side in BUF0 ----
  float* wpT   = BUF1 + 1179648;                    // [512,8192]  4,194,304
  float* weT   = wpT + 4194304;                     // [512,32]       16,384
  float* projT = weT + 16384;                       // [56,512]       28,672
  float* encT  = projT + 28672;                     // 6 x 3,145,728
  const int EB = 786432 + 262144 + 1048576 + 1048576;   // enc per-layer block
  float* decT  = BUF0 + 6000000;                    // 6 x 3,932,160
  const int DB = EB + 524288 + 262144;                  // dec per-layer block

  transp(enc_wp, wpT, 8192, 512);
  transp(dec_we, weT, 32, 512);
  transp(proj_w, projT, 512, 56);
  float *eqkvT[6], *eoutT[6], *ew1T[6], *ew2T[6];
  float *dqkvT[6], *doutT[6], *dw1T[6], *dw2T[6], *dkvT[6], *dcaoT[6];
  for (int l = 0; l < 6; ++l) {
    float* eb = encT + (size_t)l * EB;
    eqkvT[l] = eb;                 transp(esa[l].qkv, eqkvT[l], 512, 1536);
    eoutT[l] = eb + 786432;        transp(esa[l].out, eoutT[l], 512, 512);
    ew1T[l]  = eb + 1048576;       transp(eff[l].w1,  ew1T[l],  512, 2048);
    ew2T[l]  = eb + 2097152;       transp(eff[l].w2,  ew2T[l],  2048, 512);
    float* db = decT + (size_t)l * DB;
    dqkvT[l] = db;                 transp(dsa[l].qkv, dqkvT[l], 512, 1536);
    doutT[l] = db + 786432;        transp(dsa[l].out, doutT[l], 512, 512);
    dw1T[l]  = db + 1048576;       transp(dff[l].w1,  dw1T[l],  512, 2048);
    dw2T[l]  = db + 2097152;       transp(dff[l].w2,  dw2T[l],  2048, 512);
    dkvT[l]  = db + 3145728;       transp(dca[l].kv,  dkvT[l],  512, 1024);
    dcaoT[l] = db + 3670016;       transp(dca[l].out, dcaoT[l], 512, 512);
  }

  // small buffers carved past img3 region in BUF0 (below decT at 6,000,000)
  size_t off = 2000000;
  auto alloc = [&](size_t nfl) { float* p = BUF0 + off; off += nfl; return p; };
  float* xe0  = alloc(72 * 512);
  float* xe1  = alloc(72 * 512);
  float* xnE  = alloc(72 * 512);
  float* qkvE = alloc(72 * 1536);
  float* scE  = alloc(16 * 36 * 36);
  float* oE   = alloc(72 * 512);
  float* hE   = alloc(72 * 2048);
  float* memB = alloc(2 * 512);
  float* xd0  = alloc(256 * 512);
  float* xd1  = alloc(256 * 512);
  float* xn32 = alloc(256 * 32);
  float* xdn  = alloc(256 * 512);
  float* qkvD = alloc(256 * 1536);
  float* scD  = alloc(16 * 128 * 128);
  float* oD   = alloc(256 * 512);
  float* hD   = alloc(256 * 2048);
  float* kvb  = alloc(2 * 1024);
  float* cav  = alloc(2 * 512);
  float* padf = alloc(256);
  float* vT   = alloc(16 * 64 * 128);   // V transposed per layer (max n=128)

  gemmt(xn8, 8192, wpT, 512, enc_bp, nullptr, xe1, 72, 8192);   // patch embed
  lnorm(xe1, enc_ln2, xe0, 72, 512);
  k_add_pe3d<<<dim3((72 * 512 + 255) / 256), dim3(256), 0, stream>>>(xe0);

  for (int l = 0; l < 6; ++l) {
    lnorm(xe0, esa[l].ln, xnE, 72, 512);
    gemmt(xnE, 512, eqkvT[l], 1536, nullptr, nullptr, qkvE, 72, 512);
    { dim3 g(3, 1, 16); k_attn_scores<<<g, wave, 0, stream>>>(qkvE, 1536, 0, 512, 36, 0.125f, scE); }
    k_softmax<<<dim3(16 * 36), dim3(128), 0, stream>>>(scE, 36, 0, nullptr);
    k_transpose_v<<<dim3((16 * 64 * 36 + 255) / 256), dim3(256), 0, stream>>>(qkvE, vT, 36);
    { dim3 g(3, 1, 16); k_attn_out<<<g, wave, 0, stream>>>(scE, vT, 36, oE); }
    gemmt(oE, 512, eoutT[l], 512, nullptr, xe0, xe1, 72, 512);
    lnorm(xe1, eff[l].ln, xnE, 72, 512);
    gemmt(xnE, 512, ew1T[l], 2048, eff[l].b1, nullptr, hE, 72, 512);
    k_gelu<<<dim3((72 * 2048 + 255) / 256), dim3(256), 0, stream>>>(hE, 72 * 2048);
    gemmt(hE, 2048, ew2T[l], 512, eff[l].b2, xe1, xe0, 72, 2048);
  }
  k_mean_tokens<<<dim3(4), dim3(256), 0, stream>>>(xe0, memB, 36);

  // ========================= decoder =========================
  k_padflag<<<dim3(1), dim3(256), 0, stream>>>(x, padf);
  lnorm(x, dec_ln1, xn32, 256, 32);
  gemmt(xn32, 32, weT, 512, dec_be2, nullptr, xdn, 256, 32);
  lnorm(xdn, dec_ln2, xd0, 256, 512);
  k_add_pe1d<<<dim3((256 * 512 + 255) / 256), dim3(256), 0, stream>>>(xd0, 128);

  for (int l = 0; l < 6; ++l) {
    // masked self-attention
    lnorm(xd0, dsa[l].ln, xdn, 256, 512);
    gemmt(xdn, 512, dqkvT[l], 1536, nullptr, nullptr, qkvD, 256, 512);
    { dim3 g(8, 2, 16); k_attn_scores<<<g, wave, 0, stream>>>(qkvD, 1536, 0, 512, 128, 0.125f, scD); }
    k_softmax<<<dim3(16 * 128), dim3(128), 0, stream>>>(scD, 128, 1, padf);
    k_transpose_v<<<dim3((16 * 64 * 128 + 255) / 256), dim3(256), 0, stream>>>(qkvD, vT, 128);
    { dim3 g(8, 1, 16); k_attn_out<<<g, wave, 0, stream>>>(scD, vT, 128, oD); }
    gemmt(oD, 512, doutT[l], 512, nullptr, xd0, xd1, 256, 512);
    // cross-attention: keys == broadcast memory -> softmax uniform -> output = v
    gemmt(memB, 512, dkvT[l], 1024, nullptr, nullptr, kvb, 2, 512);
    gemmt(kvb + 512, 1024, dcaoT[l], 512, nullptr, nullptr, cav, 2, 512);
    k_bcast_add<<<dim3((256 * 512 + 255) / 256), dim3(256), 0, stream>>>(xd1, cav);
    // feed-forward
    lnorm(xd1, dff[l].ln, xdn, 256, 512);
    gemmt(xdn, 512, dw1T[l], 2048, dff[l].b1, nullptr, hD, 256, 512);
    k_gelu<<<dim3((256 * 2048 + 255) / 256), dim3(256), 0, stream>>>(hD, 256 * 2048);
    gemmt(hD, 2048, dw2T[l], 512, dff[l].b2, xd1, xd0, 256, 2048);
  }

  // final projection -> d_out [2,128,56]
  gemmt(xd0, 512, projT, 56, proj_b, nullptr, (float*)d_out, 256, 512);
}
